// GATEdgeAT_48155173323469
// MI455X (gfx1250) — matrix operand order repaired and checked by reading, not compiled, not run on hardware
//
#include <hip/hip_runtime.h>
#include <hip/hip_bf16.h>
#include <stdint.h>

#define NN   20000
#define EE   160000
#define FIN  30
#define ED   11
#define EMB  128
#define NH   4
#define HC   512
#define NG   1000
#define ETOT (EE + NN)

typedef __attribute__((ext_vector_type(16))) __bf16 v16bf;
typedef __attribute__((ext_vector_type(8)))  float  v8f;

struct U8x4 { uint4 lo, hi; };   // 32 bytes == v16bf

static __device__ __forceinline__ unsigned short f2bf(float f) {
  unsigned u = __float_as_uint(f);
  u += 0x7FFFu + ((u >> 16) & 1u);
  return (unsigned short)(u >> 16);
}
// monotone float <-> uint encoding for atomicMax-based segment max
static __device__ __forceinline__ unsigned encf(float f) {
  unsigned u = __float_as_uint(f);
  return (u & 0x80000000u) ? ~u : (u | 0x80000000u);
}
static __device__ __forceinline__ float decf(unsigned e) {
  unsigned u = (e & 0x80000000u) ? (e & 0x7FFFFFFFu) : ~e;
  return __uint_as_float(u);
}

// ---------------------------------------------------------------------------
// WMMA bf16 GEMM: Cout[M,Nout] = A[M,K](bf16) * W[K,Nout], W given transposed
// as Wt[Nout][K] (bf16). Optional bias[Nout].
// Block = 256 threads (8 waves), 64x64 output tile; each wave owns a 16x32
// sub-tile (two v_wmma_f32_16x16x32_bf16 per K-chunk sharing the A fragment).
// LDS double-buffered; staging = exactly one b128 global load + one b128 ds
// store per thread per tile (no branches: M-guard replaced by index clamp).
// Requires: K % 32 == 0, Nout % 64 == 0.
// ---------------------------------------------------------------------------
__global__ __launch_bounds__(256)
void gemm_bf16(const unsigned short* __restrict__ A,
               const unsigned short* __restrict__ Wt,
               float* __restrict__ Cout,
               const float* __restrict__ bias,
               int M, int K, int Nout)
{
  __shared__ alignas(16) unsigned short As[2][64][40];  // +8 pad: no conflicts
  __shared__ alignas(16) unsigned short Bs[2][64][40];

  const int tid  = threadIdx.x;
  const int row  = tid >> 2;          // 0..63 (staging row / weight col)
  const int ck   = (tid & 3) << 3;    // 0,8,16,24 (bf16 elems)
  const int mBase = blockIdx.y * 64;
  const int nBase = blockIdx.x * 64;

  const int mg = min(mBase + row, M - 1);          // clamp instead of guard
  const unsigned short* Aptr = A  + (size_t)mg * K + ck;
  const unsigned short* Bptr = Wt + (size_t)(nBase + row) * K + ck;

  // prologue: stage K-chunk 0 into buffer 0
  {
    uint4 a4 = *(const uint4*)Aptr;
    uint4 b4 = *(const uint4*)Bptr;
    *(uint4*)&As[0][row][ck] = a4;
    *(uint4*)&Bs[0][row][ck] = b4;
  }
  __syncthreads();

  const int lane = tid & 31;
  const int wv   = tid >> 5;        // wave 0..7
  const int wm   = wv & 3;          // M sub-tile 0..3
  const int wn   = wv >> 2;         // N sub-tile 0..1 (each 32 wide)
  const int hv   = lane >> 4;       // half-wave select
  const int r    = lane & 15;

  v8f acc0 = {}, acc1 = {};
  int buf = 0;
  for (int kc = 0; kc < K; kc += 32) {
    const bool nxt = (kc + 32) < K;
    uint4 anx, bnx;
    if (nxt) {                         // issue next chunk's global loads early
      anx = *(const uint4*)(Aptr + kc + 32);
      bnx = *(const uint4*)(Bptr + kc + 32);
      if (kc + 64 < K)                 // gfx1250 global_prefetch_b8
        __builtin_prefetch(Aptr + kc + 64, 0, 1);
    }

    // A fragment: half hv: elems 0-15 = A[row][hv*16 .. +15] (matches B)
    const int arow = wm * 16 + r;
    const int kb   = hv * 16;
    U8x4 ap;
    ap.lo = *(const uint4*)&As[buf][arow][kb];
    ap.hi = *(const uint4*)&As[buf][arow][kb + 8];
    v16bf av = __builtin_bit_cast(v16bf, ap);

    // B fragments: lanes 0-15 K=0..15, lanes 16-31 K=16..31, col-major LDS
    const int kb2   = hv * 16;
    const int bcol0 = wn * 32 + r;
    U8x4 bp0, bp1;
    bp0.lo = *(const uint4*)&Bs[buf][bcol0][kb2];
    bp0.hi = *(const uint4*)&Bs[buf][bcol0][kb2 + 8];
    bp1.lo = *(const uint4*)&Bs[buf][bcol0 + 16][kb2];
    bp1.hi = *(const uint4*)&Bs[buf][bcol0 + 16][kb2 + 8];
    v16bf bv0 = __builtin_bit_cast(v16bf, bp0);
    v16bf bv1 = __builtin_bit_cast(v16bf, bp1);

    acc0 = __builtin_amdgcn_wmma_f32_16x16x32_bf16(
               false, av, false, bv0, (short)0, acc0, false, false);
    acc1 = __builtin_amdgcn_wmma_f32_16x16x32_bf16(
               false, av, false, bv1, (short)0, acc1, false, false);

    if (nxt) {                         // fill the other buffer (no readers yet)
      *(uint4*)&As[buf ^ 1][row][ck] = anx;
      *(uint4*)&Bs[buf ^ 1][row][ck] = bnx;
    }
    __syncthreads();                   // s_wait_dscnt + barrier
    buf ^= 1;
  }

  // C layout: elem i -> row wm*16 + hv*8 + i, col = n0 / n0+16
  const int n0 = nBase + wn * 32 + r;
  const int n1 = n0 + 16;
  const float bv0s = bias ? bias[n0] : 0.f;
  const float bv1s = bias ? bias[n1] : 0.f;
  const int mT = mBase + wm * 16 + hv * 8;
  for (int i = 0; i < 8; ++i) {
    int m = mT + i;
    if (m < M) {
      Cout[(size_t)m * Nout + n0] = acc0[i] + bv0s;
      Cout[(size_t)m * Nout + n1] = acc1[i] + bv1s;
    }
  }
}

// ---------------------------------------------------------------------------
// Elementwise / scatter helpers
// ---------------------------------------------------------------------------
__global__ void k_zero(unsigned* p, int n) {
  int i = blockIdx.x * blockDim.x + threadIdx.x;
  if (i < n) p[i] = 0u;
}

// degree + scatter-sum of edge_attr by dst (self-loop fill 'mean')
__global__ void k_loop_acc(const int* __restrict__ ei, const float* __restrict__ ea,
                           float* __restrict__ lsum, float* __restrict__ deg) {
  int e = blockIdx.x * blockDim.x + threadIdx.x;
  if (e >= EE) return;
  int dst = ei[EE + e];
  atomicAdd(&deg[dst], 1.f);
  for (int d = 0; d < ED; ++d) atomicAdd(&lsum[dst * ED + d], ea[(size_t)e * ED + d]);
}
__global__ void k_loop_div(float* __restrict__ lsum, const float* __restrict__ deg) {
  int i = blockIdx.x * blockDim.x + threadIdx.x;
  if (i >= NN * ED) return;
  lsum[i] /= fmaxf(deg[i / ED], 1.f);
}

// weight conversions (fp32 -> bf16, transposed to [n][k])
__global__ void k_conv_lin0(const float* __restrict__ w, unsigned short* __restrict__ wt) {
  int i = blockIdx.x * blockDim.x + threadIdx.x;
  if (i >= HC * 32) return;
  int n = i >> 5, k = i & 31;
  wt[i] = (k < FIN) ? f2bf(w[(size_t)k * HC + n]) : (unsigned short)0;
}
__global__ void k_conv_lin(const float* __restrict__ w, unsigned short* __restrict__ wt) {
  int i = blockIdx.x * blockDim.x + threadIdx.x;
  if (i >= 3 * HC * EMB) return;
  int j = i / (HC * EMB), rem = i % (HC * EMB);
  int n = rem / EMB, k = rem % EMB;
  wt[i] = f2bf(w[(size_t)j * EMB * HC + (size_t)k * HC + n]);
}
__global__ void k_conv_ht(const float* __restrict__ w, unsigned short* __restrict__ wt) {
  int i = blockIdx.x * blockDim.x + threadIdx.x;
  if (i >= 4 * EMB * HC) return;
  int l = i / (EMB * HC), rem = i % (EMB * HC);
  int n = rem / HC, k = rem % HC;
  wt[i] = f2bf(w[(size_t)l * HC * EMB + (size_t)k * EMB + n]);
}
// W2[l][d][h] = sum_c edge_w[d, h*C+c] * a_e[h,c]   (collapses ee entirely)
__global__ void k_conv_w2(const float* __restrict__ e0, const float* __restrict__ a0,
                          const float* __restrict__ el, const float* __restrict__ al,
                          float* __restrict__ w2) {
  int i = blockIdx.x * blockDim.x + threadIdx.x;
  if (i >= 4 * ED * NH) return;
  int l = i / (ED * NH), rem = i % (ED * NH);
  int d = rem / NH, h = rem % NH;
  const float* ew = (l == 0) ? e0 : el + (size_t)(l - 1) * ED * HC;
  const float* ae = (l == 0) ? a0 : al + (size_t)(l - 1) * NH * EMB;
  float s = 0.f;
  for (int c = 0; c < EMB; ++c) s += ew[(size_t)d * HC + h * EMB + c] * ae[h * EMB + c];
  w2[i] = s;
}
__global__ void k_conv_x(const float* __restrict__ x, unsigned short* __restrict__ hbf) {
  int i = blockIdx.x * blockDim.x + threadIdx.x;
  if (i >= NN * 32) return;
  int n = i >> 5, k = i & 31;
  hbf[i] = (k < FIN) ? f2bf(x[(size_t)n * FIN + k]) : (unsigned short)0;
}

// per-node attention logits a_src[n,h], a_dst[n,h]
__global__ void k_node_alpha(const float* __restrict__ xs, const float* __restrict__ as,
                             const float* __restrict__ ad,
                             float* __restrict__ osrc, float* __restrict__ odst) {
  int i = blockIdx.x * blockDim.x + threadIdx.x;
  if (i >= NN * NH) return;
  int n = i >> 2, h = i & 3;
  const float* xp = xs + (size_t)n * HC + h * EMB;
  const float* sp = as + h * EMB;
  const float* dp = ad + h * EMB;
  float s = 0.f, d = 0.f;
  for (int c = 0; c < EMB; c += 4) {
    float4 v = *(const float4*)(xp + c);
    float4 a = *(const float4*)(sp + c);
    float4 b = *(const float4*)(dp + c);
    s += v.x * a.x + v.y * a.y + v.z * a.z + v.w * a.w;
    d += v.x * b.x + v.y * b.y + v.z * b.z + v.w * b.w;
  }
  osrc[i] = s; odst[i] = d;
}

// per-edge (incl self loops) alpha = leakyrelu(.,0.2); segment max via atomicMax
__global__ void k_edge_alpha(const int* __restrict__ ei, const float* __restrict__ ea,
                             const float* __restrict__ lattr,
                             const float* __restrict__ asrc, const float* __restrict__ adst,
                             const float* __restrict__ w2,
                             float* __restrict__ alpha, unsigned* __restrict__ amax) {
  int e = blockIdx.x * blockDim.x + threadIdx.x;
  if (e >= ETOT) return;
  int src, dst; const float* ap;
  if (e < EE) { src = ei[e]; dst = ei[EE + e]; ap = ea + (size_t)e * ED; }
  else        { src = dst = e - EE;            ap = lattr + (size_t)(e - EE) * ED; }
  float av[ED];
  for (int d = 0; d < ED; ++d) av[d] = ap[d];
  for (int h = 0; h < NH; ++h) {
    float t = asrc[src * NH + h] + adst[dst * NH + h];
    for (int d = 0; d < ED; ++d) t += av[d] * w2[d * NH + h];
    t = (t > 0.f) ? t : 0.2f * t;
    alpha[(size_t)e * NH + h] = t;
    atomicMax(&amax[dst * NH + h], encf(t));
  }
}

__global__ void k_edge_expden(const int* __restrict__ ei, const unsigned* __restrict__ amax,
                              float* __restrict__ alpha, float* __restrict__ den) {
  int e = blockIdx.x * blockDim.x + threadIdx.x;
  if (e >= ETOT) return;
  int dst = (e < EE) ? ei[EE + e] : (e - EE);
  for (int h = 0; h < NH; ++h) {
    float m = decf(amax[dst * NH + h]);
    float x = expf(alpha[(size_t)e * NH + h] - m);
    alpha[(size_t)e * NH + h] = x;
    atomicAdd(&den[dst * NH + h], x);
  }
}

// message scatter: out[dst] += softmax_w * xs[src]; 128 threads/edge, float4 lanes
__global__ void k_scatter(const int* __restrict__ ei, const float* __restrict__ alpha,
                          const float* __restrict__ den, const float* __restrict__ xs,
                          float* __restrict__ outacc) {
  long long gid = (long long)blockIdx.x * blockDim.x + threadIdx.x;
  if (gid >= (long long)ETOT * 128) return;
  int e = (int)(gid >> 7);
  int t = (int)(gid & 127);
  int h = t >> 5;
  int q = (t & 31) << 2;
  int src, dst;
  if (e < EE) { src = ei[e]; dst = ei[EE + e]; } else { src = dst = e - EE; }
  float w = alpha[(size_t)e * NH + h] / den[dst * NH + h];
  float4 v = *(const float4*)(xs + (size_t)src * HC + h * EMB + q);
  float* o = outacc + (size_t)dst * HC + h * EMB + q;
  atomicAdd(o + 0, v.x * w);
  atomicAdd(o + 1, v.y * w);
  atomicAdd(o + 2, v.z * w);
  atomicAdd(o + 3, v.w * w);
}

__global__ void k_bias_tanh(const float* __restrict__ acc, const float* __restrict__ bias,
                            unsigned short* __restrict__ mbf) {
  int i = blockIdx.x * blockDim.x + threadIdx.x;
  if (i >= NN * HC) return;
  mbf[i] = f2bf(tanhf(acc[i] + bias[i & (HC - 1)]));
}

__global__ void k_bn_stats(const float* __restrict__ h2, float* __restrict__ s,
                           float* __restrict__ q) {
  int c = threadIdx.x;             // 128 threads/block, 125 blocks * 160 = N
  int base = blockIdx.x * 160;
  float ls = 0.f, lq = 0.f;
  for (int i = 0; i < 160; ++i) {
    int n = base + i;
    if (n < NN) { float v = h2[(size_t)n * EMB + c]; ls += v; lq += v * v; }
  }
  atomicAdd(&s[c], ls);
  atomicAdd(&q[c], lq);
}
__global__ void k_bn_apply(const float* __restrict__ h2, const float* __restrict__ s,
                           const float* __restrict__ q, const float* __restrict__ g,
                           const float* __restrict__ b, float* __restrict__ hout,
                           unsigned short* __restrict__ hbf) {
  int i = blockIdx.x * blockDim.x + threadIdx.x;
  if (i >= NN * EMB) return;
  int c = i & (EMB - 1);
  float mu  = s[c] * (1.f / NN);
  float var = q[c] * (1.f / NN) - mu * mu;
  float v = (h2[i] - mu) * rsqrtf(var + 1e-5f) * g[c] + b[c];
  hout[i] = v;
  hbf[i]  = f2bf(v);
}

__global__ void k_pool(const float* __restrict__ h, const int* __restrict__ batch,
                       float* __restrict__ gsum, unsigned* __restrict__ gmx,
                       float* __restrict__ gcnt) {
  int i = blockIdx.x * blockDim.x + threadIdx.x;
  if (i >= NN * EMB) return;
  int n = i >> 7, c = i & 127;
  int g = batch[n];
  float v = h[i];
  atomicAdd(&gsum[g * EMB + c], v);
  atomicMax(&gmx[g * EMB + c], encf(v));
  if (c == 0) atomicAdd(&gcnt[g], 1.f);
}
__global__ void k_pool_fin(const float* __restrict__ gsum, const unsigned* __restrict__ gmx,
                           const float* __restrict__ gcnt, float* __restrict__ dout) {
  int i = blockIdx.x * blockDim.x + threadIdx.x;
  if (i >= NG * EMB) return;
  int g = i >> 7, c = i & 127;
  float cnt = gcnt[g];
  float mx = (cnt > 0.f) ? decf(gmx[i]) : 0.f;
  float mn = gsum[i] / fmaxf(cnt, 1.f);
  float* hid = dout + NG;                    // hidden follows out in d_out
  hid[(size_t)g * 2 * EMB + c]       = mx;
  hid[(size_t)g * 2 * EMB + EMB + c] = mn;
}

__global__ void k_mlp1(const float* __restrict__ hid, const float* __restrict__ w,
                       const float* __restrict__ b, float* __restrict__ o) {
  int i = blockIdx.x * blockDim.x + threadIdx.x;
  if (i >= NG * 2 * EMB) return;
  int g = i >> 8, j = i & 255;
  const float* hp = hid + (size_t)g * 2 * EMB;
  float acc = b[j];
  for (int k = 0; k < 2 * EMB; ++k) acc += hp[k] * w[(size_t)k * 2 * EMB + j];
  o[i] = fmaxf(acc, 0.f);
}
__global__ void k_mlp2(const float* __restrict__ o, const float* __restrict__ w,
                       const float* __restrict__ b, float* __restrict__ dout) {
  int g = blockIdx.x * blockDim.x + threadIdx.x;
  if (g >= NG) return;
  float acc = b[0];
  for (int j = 0; j < 2 * EMB; ++j) acc += o[(size_t)g * 2 * EMB + j] * w[j];
  dout[g] = acc;
}

// ---------------------------------------------------------------------------
extern "C" void kernel_launch(void* const* d_in, const int* in_sizes, int n_in,
                              void* d_out, int out_size, void* d_ws, size_t ws_size,
                              hipStream_t stream) {
  (void)in_sizes; (void)n_in; (void)out_size; (void)ws_size;
  const float* x        = (const float*)d_in[0];
  const int*   ei       = (const int*)  d_in[1];
  const float* ea       = (const float*)d_in[2];
  const int*   batch    = (const int*)  d_in[3];
  const float* g0_lin   = (const float*)d_in[4];
  const float* g0_edge  = (const float*)d_in[5];
  const float* g0_asrc  = (const float*)d_in[6];
  const float* g0_adst  = (const float*)d_in[7];
  const float* g0_aedge = (const float*)d_in[8];
  const float* g0_b     = (const float*)d_in[9];
  const float* g_lin    = (const float*)d_in[10];
  const float* g_edge   = (const float*)d_in[11];
  const float* g_asrc   = (const float*)d_in[12];
  const float* g_adst   = (const float*)d_in[13];
  const float* g_aedge  = (const float*)d_in[14];
  const float* g_b      = (const float*)d_in[15];
  const float* ht_w     = (const float*)d_in[16];
  const float* ht_b     = (const float*)d_in[17];
  const float* bn_g     = (const float*)d_in[18];
  const float* bn_b     = (const float*)d_in[19];
  const float* out1_w   = (const float*)d_in[20];
  const float* out1_b   = (const float*)d_in[21];
  const float* out2_w   = (const float*)d_in[22];
  const float* out2_b   = (const float*)d_in[23];
  float* dout = (float*)d_out;

  char* ws = (char*)d_ws;
  size_t off = 0;
  auto alloc = [&](size_t bytes) -> char* {
    char* p = ws + off;
    off += (bytes + 255) & ~(size_t)255;
    return p;
  };
  float*          xs    = (float*)          alloc((size_t)NN * HC * 4);
  float*          gacc  = (float*)          alloc((size_t)NN * HC * 4);
  unsigned short* mbf   = (unsigned short*) alloc((size_t)NN * HC * 2);
  unsigned short* hbf   = (unsigned short*) alloc((size_t)NN * EMB * 2);
  float*          hcur  = (float*)          alloc((size_t)NN * EMB * 4);
  float*          alpha = (float*)          alloc((size_t)ETOT * NH * 4);
  float*          asrc  = (float*)          alloc((size_t)NN * NH * 4);
  float*          adst  = (float*)          alloc((size_t)NN * NH * 4);
  unsigned*       amax  = (unsigned*)       alloc((size_t)NN * NH * 4);
  float*          den   = (float*)          alloc((size_t)NN * NH * 4);
  float*          lattr = (float*)          alloc((size_t)NN * ED * 4);
  float*          deg   = (float*)          alloc((size_t)NN * 4);
  unsigned short* wt0   = (unsigned short*) alloc((size_t)HC * 32 * 2);
  unsigned short* wtl   = (unsigned short*) alloc((size_t)3 * HC * EMB * 2);
  unsigned short* wth   = (unsigned short*) alloc((size_t)4 * EMB * HC * 2);
  float*          w2    = (float*)          alloc((size_t)4 * ED * NH * 4);
  float*          bns   = (float*)          alloc(EMB * 4);
  float*          bnq   = (float*)          alloc(EMB * 4);
  float*          gsum  = (float*)          alloc((size_t)NG * EMB * 4);
  unsigned*       gmx   = (unsigned*)       alloc((size_t)NG * EMB * 4);
  float*          gcnt  = (float*)          alloc((size_t)NG * 4);
  float*          obuf  = (float*)          alloc((size_t)NG * 2 * EMB * 4);

#define GR(n) dim3(((n) + 255) / 256), dim3(256), 0, stream

  // self-loop attrs, weight conversions (deterministic every call)
  k_zero<<<GR(NN)>>>((unsigned*)deg, NN);
  k_zero<<<GR(NN * ED)>>>((unsigned*)lattr, NN * ED);
  k_loop_acc<<<GR(EE)>>>(ei, ea, lattr, deg);
  k_loop_div<<<GR(NN * ED)>>>(lattr, deg);
  k_conv_lin0<<<GR(HC * 32)>>>(g0_lin, wt0);
  k_conv_lin<<<GR(3 * HC * EMB)>>>(g_lin, wtl);
  k_conv_ht<<<GR(4 * EMB * HC)>>>(ht_w, wth);
  k_conv_w2<<<GR(4 * ED * NH)>>>(g0_edge, g0_aedge, g_edge, g_aedge, w2);
  k_conv_x<<<GR(NN * 32)>>>(x, hbf);

  for (int l = 0; l < 4; ++l) {
    const int K = (l == 0) ? 32 : EMB;
    const unsigned short* lin = (l == 0) ? wt0 : wtl + (size_t)(l - 1) * HC * EMB;
    const float* a_s  = (l == 0) ? g0_asrc : g_asrc + (size_t)(l - 1) * NH * EMB;
    const float* a_d  = (l == 0) ? g0_adst : g_adst + (size_t)(l - 1) * NH * EMB;
    const float* bias = (l == 0) ? g0_b    : g_b    + (size_t)(l - 1) * HC;

    // xs = h @ lin_w   [N,K]x[K,512]  (WMMA)
    gemm_bf16<<<dim3(HC / 64, (NN + 63) / 64), dim3(256), 0, stream>>>(
        hbf, lin, xs, nullptr, NN, K, HC);

    k_node_alpha<<<GR(NN * NH)>>>(xs, a_s, a_d, asrc, adst);
    k_zero<<<GR(NN * NH)>>>(amax, NN * NH);
    k_zero<<<GR(NN * NH)>>>((unsigned*)den, NN * NH);
    k_zero<<<GR(NN * HC)>>>((unsigned*)gacc, NN * HC);
    k_edge_alpha<<<GR(ETOT)>>>(ei, ea, lattr, asrc, adst, w2 + (size_t)l * ED * NH,
                               alpha, amax);
    k_edge_expden<<<GR(ETOT)>>>(ei, amax, alpha, den);
    k_scatter<<<dim3((unsigned)(((long long)ETOT * 128 + 255) / 256)), dim3(256), 0, stream>>>(
        ei, alpha, den, xs, gacc);
    k_bias_tanh<<<GR(NN * HC)>>>(gacc, bias, mbf);

    // h2 = tanh(out) @ ht_w[l] + ht_b[l]  [N,512]x[512,128]  (WMMA), into xs
    gemm_bf16<<<dim3(EMB / 64, (NN + 63) / 64), dim3(256), 0, stream>>>(
        mbf, wth + (size_t)l * EMB * HC, xs, ht_b + (size_t)l * EMB, NN, HC, EMB);

    k_zero<<<GR(EMB)>>>((unsigned*)bns, EMB);
    k_zero<<<GR(EMB)>>>((unsigned*)bnq, EMB);
    k_bn_stats<<<dim3(125), dim3(128), 0, stream>>>(xs, bns, bnq);
    k_bn_apply<<<GR(NN * EMB)>>>(xs, bns, bnq, bn_g + (size_t)l * EMB,
                                 bn_b + (size_t)l * EMB, hcur, hbf);
  }

  // readout
  k_zero<<<GR(NG * EMB)>>>((unsigned*)gsum, NG * EMB);
  k_zero<<<GR(NG * EMB)>>>(gmx, NG * EMB);
  k_zero<<<GR(NG)>>>((unsigned*)gcnt, NG);
  k_pool<<<GR(NN * EMB)>>>(hcur, batch, gsum, gmx, gcnt);
  k_pool_fin<<<GR(NG * EMB)>>>(gsum, gmx, gcnt, dout);
  k_mlp1<<<GR(NG * 2 * EMB)>>>(dout + NG, out1_w, out1_b, obuf);
  k_mlp2<<<GR(NG)>>>(obuf, out2_w, out2_b, dout);
#undef GR
}
